// AttentionHead_81217831567433
// MI455X (gfx1250) — compile-verified, matrix operand order checked by reading, not run on hardware
//
#include <hip/hip_runtime.h>
#include <hip/hip_bf16.h>

typedef __attribute__((ext_vector_type(16))) __bf16 v16bf;
typedef __attribute__((ext_vector_type(8)))  float  v8f;

#define D_MODEL 1024
#define D_HEAD  64
#define SEQ_T   4096
#define BATCH   4
#define BT_ROWS (BATCH * SEQ_T)

union V16U { unsigned u[8]; uint4 q[2]; v16bf v; };

// native f32->bf16 (RNE): v_cvt_pk_bf16_f32 on gfx1250
__device__ __forceinline__ __bf16 f2bf(float f) { return (__bf16)f; }
__device__ __forceinline__ unsigned pkbf2(float lo, float hi) {
  union { __bf16 b[2]; unsigned u; } t;
  t.b[0] = (__bf16)lo; t.b[1] = (__bf16)hi;
  return t.u;
}
__device__ __forceinline__ float fexp2(float x) {
  return __builtin_amdgcn_exp2f(x);     // v_exp_f32 (trans, co-executes w/ WMMA)
}
__device__ __forceinline__ v8f wmma_bf16(v16bf a, v16bf b, v8f c) {
  return __builtin_amdgcn_wmma_f32_16x16x32_bf16(false, a, false, b, (short)0, c,
                                                 false, false);
}
// two 16B loads (runs are 8 elems at +0 and 8 elems at +16 elements)
__device__ __forceinline__ v16bf ld_frag16(const __bf16* p) {
  V16U t;
  t.q[0] = *(const uint4*)p;
  t.q[1] = *(const uint4*)(p + 16);
  return t.v;
}

// ---------------------------------------------------------------------------
// Kernel 0: pack Wq/Wk/Wv (f32 [1024][64]) into fragment-major bf16:
//   Wp[frag][lane][e], frag = chunk*12 + mat*4 + ht,  (512 elems per frag)
//   lane L: n = L&15, kb8 = (L>>4)*8;  elem e: k = (e<8?0:16)+kb8+(e&7)
// ---------------------------------------------------------------------------
__global__ __launch_bounds__(256) void w_pack_kernel(
    const float* __restrict__ Wq, const float* __restrict__ Wk,
    const float* __restrict__ Wv, __bf16* __restrict__ Wp)
{
  const int t     = blockIdx.x * 256 + threadIdx.x;   // 0..24575
  const int dst8  = t << 3;                           // 8 elems per thread
  const int frag  = dst8 >> 9;
  const int within= dst8 & 511;
  const int chunk = frag / 12;
  const int mat   = (frag % 12) >> 2;
  const int ht    = frag & 3;
  const int L     = within >> 4;
  const int e0    = within & 15;                      // 0 or 8
  const int n     = L & 15;
  const int kgrp  = (L >> 4) | ((e0 >> 3) << 1);
  const int kbase = chunk * 32 + kgrp * 8;
  const int h     = ht * 16 + n;
  const float* W  = (mat == 0) ? Wq : (mat == 1) ? Wk : Wv;

  float f[8];
#pragma unroll
  for (int i = 0; i < 8; ++i)
    f[i] = W[(long)(kbase + i) * D_HEAD + h];
  uint4 p;
  p.x = pkbf2(f[0], f[1]);
  p.y = pkbf2(f[2], f[3]);
  p.z = pkbf2(f[4], f[5]);
  p.w = pkbf2(f[6], f[7]);
  *(uint4*)&Wp[dst8] = p;
}

// ---------------------------------------------------------------------------
// Kernel 1: QKV projection, no LDS / no barriers, double-buffered.
// 256 threads (8 waves); block = 128 t-rows; wave = 16 rows, 12 acc tiles.
// x (HBM) loads for chunk+1 issued before the 12 WMMAs of chunk; W B-frags
// two-buffer rotated so load(f+1) is in flight during wmma(f).
// ---------------------------------------------------------------------------
__global__ __launch_bounds__(256) void qkv_proj_kernel(
    const float* __restrict__ x, const __bf16* __restrict__ Wp,
    __bf16* __restrict__ Qo, __bf16* __restrict__ Ko, __bf16* __restrict__ Vt)
{
  const int tid  = threadIdx.x;
  const int wave = tid >> 5;
  const int lane = tid & 31;
  const int m    = lane & 15;
  const int kb8  = (lane >> 4) << 3;
  const long row0 = (long)blockIdx.x * 128;
  const long arow = row0 + wave * 16 + m;
  const float* xrow = x + arow * D_MODEL;

  v8f acc[12];
#pragma unroll
  for (int i = 0; i < 12; ++i)
#pragma unroll
    for (int j = 0; j < 8; ++j) acc[i][j] = 0.0f;

  // prologue: x loads for chunk 0
  float4 xf0, xf1, xf2, xf3;
  {
    const float* p0 = xrow + kb8;
    xf0 = *(const float4*)(p0);
    xf1 = *(const float4*)(p0 + 4);
    xf2 = *(const float4*)(p0 + 16);
    xf3 = *(const float4*)(p0 + 20);
  }

  for (int chunk = 0; chunk < 32; ++chunk) {
    // issue next chunk's x loads first (HBM latency hidden under 12 WMMAs)
    float4 xn0, xn1, xn2, xn3;
    if (chunk + 1 < 32) {
      const float* pn = xrow + ((chunk + 1) << 5) + kb8;
      xn0 = *(const float4*)(pn);
      xn1 = *(const float4*)(pn + 4);
      xn2 = *(const float4*)(pn + 16);
      xn3 = *(const float4*)(pn + 20);
      __builtin_prefetch(pn + 32, 0, 1);   // global_prefetch_b8 (chunk+2)
    }
    // A fragment from current chunk's x
    V16U av;
    av.u[0] = pkbf2(xf0.x, xf0.y); av.u[1] = pkbf2(xf0.z, xf0.w);
    av.u[2] = pkbf2(xf1.x, xf1.y); av.u[3] = pkbf2(xf1.z, xf1.w);
    av.u[4] = pkbf2(xf2.x, xf2.y); av.u[5] = pkbf2(xf2.z, xf2.w);
    av.u[6] = pkbf2(xf3.x, xf3.y); av.u[7] = pkbf2(xf3.z, xf3.w);
    const v16bf a = av.v;

    // 12 B fragments, two-buffer rotation: load(f+1) in flight during wmma(f)
    const __bf16* wb = Wp + ((long)chunk * 12) * 512 + lane * 16;
    V16U bva, bvb;
    bva.q[0] = *(const uint4*)(wb);
    bva.q[1] = *(const uint4*)(wb + 8);
#pragma unroll
    for (int f = 0; f < 12; ++f) {
      V16U& cur = (f & 1) ? bvb : bva;
      V16U& nxt = (f & 1) ? bva : bvb;
      if (f + 1 < 12) {
        nxt.q[0] = *(const uint4*)(wb + (f + 1) * 512);
        nxt.q[1] = *(const uint4*)(wb + (f + 1) * 512 + 8);
      }
      acc[f] = wmma_bf16(a, cur.v, acc[f]);
    }
    xf0 = xn0; xf1 = xn1; xf2 = xn2; xf3 = xn3;
  }

  // epilogue.  C layout: vgpr r -> row mh8+r, col n=m.
  const int mh8 = (lane >> 4) << 3;
  const long b  = row0 >> 12;                      // batch
  const long t0 = (row0 & 4095) + wave * 16 + mh8; // t within batch
#pragma unroll
  for (int mat = 0; mat < 2; ++mat) {
    __bf16* outp = mat ? Ko : Qo;
#pragma unroll
    for (int ht = 0; ht < 4; ++ht) {
      v8f c = acc[mat * 4 + ht];
#pragma unroll
      for (int r = 0; r < 8; ++r)
        outp[(row0 + wave * 16 + mh8 + r) * D_HEAD + (ht << 4) + m] = f2bf(c[r]);
    }
  }
#pragma unroll
  for (int ht = 0; ht < 4; ++ht) {                 // V transposed, packed b128
    v8f c = acc[8 + ht];
    uint4 p;
    p.x = pkbf2(c[0], c[1]); p.y = pkbf2(c[2], c[3]);
    p.z = pkbf2(c[4], c[5]); p.w = pkbf2(c[6], c[7]);
    *(uint4*)&Vt[(b * D_HEAD + (ht << 4) + m) * SEQ_T + t0] = p;
  }
}

// ---------------------------------------------------------------------------
// Kernel 2: causal flash attention, online softmax (base-2 domain).
// 128 threads = 4 waves, 64 q-rows per block, 32 keys per step (P maps onto
// one bf16 A-fragment).  No barriers: all K/V fragment loads for a step are
// issued up front so their L2 latency overlaps the S-WMMAs and softmax.
// ---------------------------------------------------------------------------
__global__ __launch_bounds__(128) void flash_attn_kernel(
    const __bf16* __restrict__ Q, const __bf16* __restrict__ K,
    const __bf16* __restrict__ Vt, float* __restrict__ out)
{
  __shared__ __align__(16) __bf16 sP[4][16 * 32];

  const int tid  = threadIdx.x;
  const int wave = tid >> 5;
  const int lane = tid & 31;
  const int m    = lane & 15;
  const int kb8  = (lane >> 4) << 3;
  const int mh8  = kb8;
  const long bidx = blockIdx.y;
  const long base = bidx * SEQ_T;
  const int q0   = blockIdx.x * 64;
  const int qt0  = q0 + wave * 16;

  // Q fragments: dh split into two K=32 chunks
  v16bf qa[2];
#pragma unroll
  for (int c = 0; c < 2; ++c)
    qa[c] = ld_frag16(Q + (base + qt0 + m) * D_HEAD + (c << 5) + kb8);

  v8f o[4];
#pragma unroll
  for (int i = 0; i < 4; ++i)
#pragma unroll
    for (int j = 0; j < 8; ++j) o[i][j] = 0.0f;
  float mst[8], lst[8];
#pragma unroll
  for (int r = 0; r < 8; ++r) { mst[r] = -1e30f; lst[r] = 0.0f; }

  // D_MODEL^-0.5 * log2(e): softmax computed in exp2 domain (exact rescale)
  const float scale2 = 0.03125f * 1.4426950408889634f;

  for (int k0 = 0; k0 <= qt0 + 15; k0 += 32) {     // per-wave causal bound
    // issue all fragment loads for this step up front (one clause)
    v16bf kb00 = ld_frag16(K + (base + k0 + m)      * D_HEAD + kb8);
    v16bf kb01 = ld_frag16(K + (base + k0 + m)      * D_HEAD + 32 + kb8);
    v16bf kb10 = ld_frag16(K + (base + k0 + 16 + m) * D_HEAD + kb8);
    v16bf kb11 = ld_frag16(K + (base + k0 + 16 + m) * D_HEAD + 32 + kb8);
    v16bf vb[4];
#pragma unroll
    for (int ht = 0; ht < 4; ++ht)
      vb[ht] = ld_frag16(Vt + (bidx * D_HEAD + (ht << 4) + m) * SEQ_T + k0 + kb8);

    // S = Q K^T for 32 keys (two 16-key tiles)
    v8f s0, s1;
#pragma unroll
    for (int j = 0; j < 8; ++j) { s0[j] = 0.0f; s1[j] = 0.0f; }
    s0 = wmma_bf16(qa[0], kb00, s0);
    s0 = wmma_bf16(qa[1], kb01, s0);
    s1 = wmma_bf16(qa[0], kb10, s1);
    s1 = wmma_bf16(qa[1], kb11, s1);

    // online softmax over these 32 keys
#pragma unroll
    for (int r = 0; r < 8; ++r) {
      const int qrow = qt0 + mh8 + r;
      float t0 = ((k0 + m)      <= qrow) ? s0[r] * scale2 : -1e30f;
      float t1 = ((k0 + m + 16) <= qrow) ? s1[r] * scale2 : -1e30f;
      float mx = fmaxf(t0, t1);
#pragma unroll
      for (int off = 1; off < 16; off <<= 1)
        mx = fmaxf(mx, __shfl_xor(mx, off, 32));
      float mnew = fmaxf(mst[r], mx);
      float corr = fexp2(mst[r] - mnew);
      float p0 = fexp2(t0 - mnew);
      float p1 = fexp2(t1 - mnew);
      float rs = p0 + p1;
#pragma unroll
      for (int off = 1; off < 16; off <<= 1)
        rs += __shfl_xor(rs, off, 32);
      mst[r] = mnew;
      lst[r] = lst[r] * corr + rs;
#pragma unroll
      for (int ht = 0; ht < 4; ++ht) o[ht][r] *= corr;
      sP[wave][(mh8 + r) * 32 + m]      = f2bf(p0);
      sP[wave][(mh8 + r) * 32 + m + 16] = f2bf(p1);
    }
    // P as A-fragment (same-wave LDS is in-order; 2x ds_load_b128)
    v16bf pa = ld_frag16(&sP[wave][m * 32 + kb8]);
    // O += P V
#pragma unroll
    for (int ht = 0; ht < 4; ++ht)
      o[ht] = wmma_bf16(pa, vb[ht], o[ht]);
  }

  // epilogue: normalize by row sums, store f32
#pragma unroll
  for (int ht = 0; ht < 4; ++ht)
#pragma unroll
    for (int r = 0; r < 8; ++r) {
      long row = base + qt0 + mh8 + r;
      out[row * D_HEAD + (ht << 4) + m] = o[ht][r] / lst[r];
    }
}

extern "C" void kernel_launch(void* const* d_in, const int* in_sizes, int n_in,
                              void* d_out, int out_size, void* d_ws, size_t ws_size,
                              hipStream_t stream) {
  const float* x  = (const float*)d_in[0];
  const float* Wq = (const float*)d_in[1];
  const float* Wk = (const float*)d_in[2];
  const float* Wv = (const float*)d_in[3];

  __bf16* Qw  = (__bf16*)d_ws;                      // 2 MB
  __bf16* Kw  = Qw + (size_t)BT_ROWS * D_HEAD;      // 2 MB
  __bf16* Vtw = Kw + (size_t)BT_ROWS * D_HEAD;      // 2 MB (transposed layout)
  __bf16* Wp  = Vtw + (size_t)BT_ROWS * D_HEAD;     // 384 KB packed weights

  w_pack_kernel<<<dim3(96), dim3(256), 0, stream>>>(Wq, Wk, Wv, Wp);
  qkv_proj_kernel<<<dim3(BT_ROWS / 128), dim3(256), 0, stream>>>(x, Wp, Qw, Kw, Vtw);
  flash_attn_kernel<<<dim3(SEQ_T / 64, BATCH), dim3(128), 0, stream>>>(
      Qw, Kw, Vtw, (float*)d_out);

  (void)in_sizes; (void)n_in; (void)out_size; (void)ws_size;
}
